// LinearAttention_21878563406364
// MI455X (gfx1250) — matrix operand order repaired and checked by reading, not compiled, not run on hardware
//
#include <hip/hip_runtime.h>
#include <hip/hip_bf16.h>
#include <math.h>

#define N_   4
#define L_   4096
#define S_   4096
#define H_   8
#define D_   64
#define DV_  64
#define EPSF 1e-6f
#define CHUNKS 16
#define SC   (S_ / CHUNKS)   // 256 source positions per chunk
#define NKS  (SC / 32)       // 8 k-steps per chunk
#define LBLK 128             // query rows per block in phase 2

typedef __attribute__((ext_vector_type(16))) __bf16 v16bf;
typedef __attribute__((ext_vector_type(8)))  __bf16 v8bf;
typedef __attribute__((ext_vector_type(8)))  float  v8f;

// elu(x) + 1  ==  x+1 (x>0)  else exp(x)
__device__ __forceinline__ float featmap(float x) {
    return x > 0.0f ? x + 1.0f : __expf(x);
}

// Build a 16-element bf16 fragment from two contiguous 8-element LDS runs
// (each run is one ds_load_b128).
__device__ __forceinline__ v16bf frag_from_lds(const __bf16* p0, const __bf16* p1) {
    v8bf a = *(const v8bf*)p0;
    v8bf b = *(const v8bf*)p1;
    return __builtin_shufflevector(a, b, 0,1,2,3,4,5,6,7,8,9,10,11,12,13,14,15);
}

// ---------------------------------------------------------------------------
// Phase 1: per (n,h,chunk) accumulate KV_partial[64x64] and ksum_partial[64]
// KV[d,v] = sum_s K_feat[s,d] * (V[s,v]/S)
// ---------------------------------------------------------------------------
__global__ __launch_bounds__(256) void kv_build(const float* __restrict__ Kg,
                                                const float* __restrict__ Vg,
                                                float* __restrict__ KVpart,
                                                float* __restrict__ KSpart) {
    // double-buffered transposed tiles: [d or v][s_local], s contiguous
    __shared__ alignas(16) __bf16 ldsK[2][D_  * 32];
    __shared__ alignas(16) __bf16 ldsV[2][DV_ * 32];
    __shared__ float red[256];

    const int t  = threadIdx.x;
    const int c  = blockIdx.x % CHUNKS;
    const int nh = blockIdx.x / CHUNKS;
    const int n  = nh / H_, h = nh % H_;
    const int s0 = c * SC;
    const int d  = t & 63;       // staged feature dim (coalesced across lanes)
    const int sg = t >> 6;       // 0..3: this thread owns s = sg*8 .. sg*8+7
    const float inv_len = 1.0f / (float)S_;

    const int wv   = t >> 5;           // wave 0..7
    const int lane = t & 31;
    const int hi   = lane >> 4;
    const int lm   = lane & 15;
    const int tr   = wv >> 1;          // KV tile row (d)  0..3
    const int tc0  = (wv & 1) * 2;     // KV tile col base (v): 0 or 2

    v8f c0 = {}; v8f c1 = {};
    float ks_acc = 0.0f;

    // stage one 32x64 tile of feature-mapped K and scaled V (transposed, bf16)
    // thread writes 8 *contiguous* s-positions -> single ds_store_b128 per array
    auto stage = [&](int ks) {
        const int buf = ks & 1;
        const int sb  = s0 + ks * 32 + sg * 8;
        v8bf kb, vb;
        #pragma unroll
        for (int i = 0; i < 8; ++i) {
            const size_t base = ((size_t)(n * S_ + sb + i) * H_ + h) * D_ + d;
            float kf = featmap(Kg[base]);
            ks_acc += kf;                        // f32 ksum partial (exact map)
            kb[i] = (__bf16)kf;
            vb[i] = (__bf16)(Vg[base] * inv_len);
        }
        *(v8bf*)&ldsK[buf][d * 32 + sg * 8] = kb;
        *(v8bf*)&ldsV[buf][d * 32 + sg * 8] = vb;
    };

    stage(0);
    for (int ks = 0; ks < NKS; ++ks) {
        __syncthreads();                 // prev compute done, current tile ready
        if (ks + 1 < NKS) stage(ks + 1); // overlap next-tile loads with WMMA
        const __bf16* bK = ldsK[ks & 1];
        const __bf16* bV = ldsV[ks & 1];

        // A = K^T tile (M=d local, K=s): lane row = tr*16+lm
        //   s = hi*16 + e
        v16bf a  = frag_from_lds(&bK[(tr * 16 + lm) * 32 + hi * 16],
                                 &bK[(tr * 16 + lm) * 32 + hi * 16 + 8]);
        // B = V tile (K=s, N=v): lane col = tc*16+lm ; s = hi*16 + e
        v16bf b0 = frag_from_lds(&bV[((tc0    ) * 16 + lm) * 32 + hi * 16],
                                 &bV[((tc0    ) * 16 + lm) * 32 + hi * 16 + 8]);
        v16bf b1 = frag_from_lds(&bV[((tc0 + 1) * 16 + lm) * 32 + hi * 16],
                                 &bV[((tc0 + 1) * 16 + lm) * 32 + hi * 16 + 8]);
        c0 = __builtin_amdgcn_wmma_f32_16x16x32_bf16(false, a, false, b0,
                                                     (short)0, c0, false, false);
        c1 = __builtin_amdgcn_wmma_f32_16x16x32_bf16(false, a, false, b1,
                                                     (short)0, c1, false, false);
    }

    // write KV partial tiles (C layout: VGPR r -> M = r + hi*8, N = lm)
    float* kvb = KVpart + ((size_t)nh * CHUNKS + c) * (D_ * DV_);
    #pragma unroll
    for (int r = 0; r < 8; ++r) {
        const int m = tr * 16 + r + hi * 8;
        kvb[m * DV_ + (tc0    ) * 16 + lm] = c0[r];
        kvb[m * DV_ + (tc0 + 1) * 16 + lm] = c1[r];
    }

    // ksum partial: 4 threads per d -> tree via LDS
    __syncthreads();
    red[t] = ks_acc;
    __syncthreads();
    if (t < D_) {
        float s = red[t] + red[t + 64] + red[t + 128] + red[t + 192];
        KSpart[((size_t)nh * CHUNKS + c) * D_ + t] = s;
    }
}

// ---------------------------------------------------------------------------
// Phase 2: out[l,v] = S * Z[l] * sum_d Q_feat[l,d] * KV[d,v]
// ---------------------------------------------------------------------------
__global__ __launch_bounds__(256) void attn_out(const float* __restrict__ Qg,
                                                const float* __restrict__ KVpart,
                                                const float* __restrict__ KSpart,
                                                float* __restrict__ Out) {
    __shared__ alignas(16) __bf16 ldsKV[DV_ * D_];  // [v][d], d contiguous
    __shared__ float ksum[D_];
    __shared__ float zrow[LBLK];

    const int t    = threadIdx.x;
    const int nblk = L_ / LBLK;
    const int lb   = blockIdx.x % nblk;
    const int nh   = blockIdx.x / nblk;
    const int n    = nh / H_, h = nh % H_;
    const int l0   = lb * LBLK;

    // reduce the KV partials straight into LDS (transposed, bf16);
    // thread owns v = t&63 and 16 contiguous d -> two ds_store_b128
    const float* kvb = KVpart + (size_t)nh * CHUNKS * (D_ * DV_);
    {
        const int v  = t & 63;
        const int g  = t >> 6;        // 0..3 -> d block of 16
        v8bf pk[2];
        #pragma unroll
        for (int j = 0; j < 16; ++j) {
            const int dd = g * 16 + j;
            float s = 0.0f;
            #pragma unroll
            for (int c = 0; c < CHUNKS; ++c)
                s += kvb[c * (D_ * DV_) + dd * DV_ + v];
            pk[j >> 3][j & 7] = (__bf16)s;
        }
        *(v8bf*)&ldsKV[v * D_ + g * 16]     = pk[0];
        *(v8bf*)&ldsKV[v * D_ + g * 16 + 8] = pk[1];
    }
    if (t < D_) {
        float s = 0.0f;
        #pragma unroll
        for (int c = 0; c < CHUNKS; ++c)
            s += KSpart[((size_t)nh * CHUNKS + c) * D_ + t];
        ksum[t] = s;
    }
    __syncthreads();

    // Z per row, full f32 from global Q
    if (t < LBLK) {
        const float* qr = Qg + ((size_t)(n * L_ + l0 + t) * H_ + h) * D_;
        float acc = 0.0f;
        #pragma unroll
        for (int dd = 0; dd < D_; ++dd) acc += featmap(qr[dd]) * ksum[dd];
        zrow[t] = 1.0f / (acc + EPSF);
    }
    __syncthreads();

    const int wv = t >> 5, lane = t & 31, hi = lane >> 4, lm = lane & 15;
    const int lrow = l0 + wv * 16 + lm;
    const float* qr = Qg + ((size_t)(n * L_ + lrow) * H_ + h) * D_;

    v8f acc[4] = {{}, {}, {}, {}};
    #pragma unroll
    for (int kk = 0; kk < 2; ++kk) {        // k0 = 0, 32
        const int k0 = kk * 32;
        // A fragment straight from global Q (feature map applied on the fly)
        v16bf a;
        #pragma unroll
        for (int i = 0; i < 8; ++i) {
            a[i]     = (__bf16)featmap(qr[k0 + hi * 16 + i]);
            a[i + 8] = (__bf16)featmap(qr[k0 + hi * 16 + 8 + i]);
        }
        #pragma unroll
        for (int tc = 0; tc < 4; ++tc) {
            v16bf b = frag_from_lds(&ldsKV[(tc * 16 + lm) * D_ + k0 + hi * 16],
                                    &ldsKV[(tc * 16 + lm) * D_ + k0 + hi * 16 + 8]);
            acc[tc] = __builtin_amdgcn_wmma_f32_16x16x32_bf16(false, a, false, b,
                                                              (short)0, acc[tc],
                                                              false, false);
        }
    }

    #pragma unroll
    for (int tc = 0; tc < 4; ++tc) {
        #pragma unroll
        for (int r = 0; r < 8; ++r) {
            const int lr  = wv * 16 + r + hi * 8;
            const float o = acc[tc][r] * zrow[lr] * (float)S_;
            Out[((size_t)(n * L_ + l0 + lr) * H_ + h) * DV_ + tc * 16 + lm] = o;
        }
    }
}

extern "C" void kernel_launch(void* const* d_in, const int* in_sizes, int n_in,
                              void* d_out, int out_size, void* d_ws, size_t ws_size,
                              hipStream_t stream) {
    const float* Q = (const float*)d_in[0];
    const float* K = (const float*)d_in[1];
    const float* V = (const float*)d_in[2];
    float* Out = (float*)d_out;

    // workspace: KV partials [N*H*CHUNKS][64*64], ksum partials [N*H*CHUNKS][64]
    float* KVpart = (float*)d_ws;
    float* KSpart = KVpart + (size_t)N_ * H_ * CHUNKS * D_ * DV_;

    hipLaunchKernelGGL(kv_build, dim3(N_ * H_ * CHUNKS), dim3(256), 0, stream,
                       K, V, KVpart, KSpart);
    hipLaunchKernelGGL(attn_out, dim3(N_ * H_ * (L_ / LBLK)), dim3(256), 0, stream,
                       Q, KVpart, KSpart, Out);
}
